// AttenPropagation_54631984005671
// MI455X (gfx1250) — matrix operand order repaired and checked by reading, not compiled, not run on hardware
//
#include <hip/hip_runtime.h>
#include <hip/hip_bf16.h>

typedef __attribute__((ext_vector_type(16))) __bf16 v16bf;
typedef __attribute__((ext_vector_type(8)))  float  v8f;

#define NPIX 5120
#define MKEY 4096
#define DCH  64
#define HD   16
#define NH   4
#define H2   128

union BfVec {
  v16bf v;
  uint4 q[2];
  __bf16 h[16];
};

// ---------------------------------------------------------------------------
// 64-byte zero pad used as the B-operand source for the hi half-wave
// (K = 16..31 zero padding of the head-dim contraction).
// ---------------------------------------------------------------------------
__global__ __launch_bounds__(32) void init_zpad(__bf16* __restrict__ z)
{
  z[threadIdx.x] = (__bf16)0.f;
}

// ---------------------------------------------------------------------------
// Projection: out[o][n] = sum_c W[o][c] X[c][n] + b[o], written as bf16 in a
// WMMA-friendly per-head layout. o = d*4 + h (torch reshape (hd, heads)).
//   transposed==0 : out[h][n][d]   (contig d, 16 per row)  -> A / score-B side
//   transposed==1 : out[h][d][m]   (contig m)              -> PV B side
// ---------------------------------------------------------------------------
__global__ __launch_bounds__(256) void proj_kernel(
    const float* __restrict__ X, const float* __restrict__ W,
    const float* __restrict__ b, __bf16* __restrict__ out,
    int Ntot, int transposed)
{
  int g = blockIdx.x * 256 + threadIdx.x;
  if (g >= DCH * Ntot) return;
  int n = g % Ntot;
  int o = g / Ntot;
  float acc = b[o];
  #pragma unroll
  for (int c = 0; c < DCH; ++c)
    acc += W[o * DCH + c] * X[(size_t)c * Ntot + n];
  int d = o >> 2, hh = o & 3;
  if (transposed) out[((size_t)hh * HD + d) * Ntot + n] = (__bf16)acc;
  else            out[((size_t)hh * Ntot + n) * HD + d] = (__bf16)acc;
}

// ---------------------------------------------------------------------------
// Flash attention: one wave per (head, 16-row n-tile).
// Pass A: row-max of scores (WMMA + 8 fmax per 16-col chunk).
// Pass B: unnormalized p = exp(s - mx) -> PV WMMA accumulate + rowsum.
// Final:  X *= 1/rowsum (normalization deferred out of the loop).
// K-tile loads are unconditional: lo lanes stride through kh, hi lanes have
// stride 0 into a 64B zero pad (K=16..31 padding) -> no divergence, no
// per-iteration zero re-materialization.
// ---------------------------------------------------------------------------
__global__ __launch_bounds__(32) void attn_flash(
    const __bf16* __restrict__ qh, const __bf16* __restrict__ kh,
    const __bf16* __restrict__ vh, const __bf16* __restrict__ zpad,
    float* __restrict__ xout)
{
  const int lane = threadIdx.x;
  const int head = blockIdx.x / (NPIX / 16);
  const int n0   = (blockIdx.x % (NPIX / 16)) * 16;
  const bool lo  = lane < 16;
  const int  l16 = lane & 15;

  __shared__ __align__(16) __bf16 plds[16][32];

  // --- load Q as WMMA A operand (16n x 32K, K=16..31 zero) ---
  BfVec aq;
  {
    const __bf16* qp = qh + ((size_t)head * NPIX + (n0 + l16)) * HD;
    aq.q[0] = lo ? *(const uint4*)qp : uint4{0u, 0u, 0u, 0u};
    aq.q[1] = lo ? *(const uint4*)(qp + 8) : uint4{0u, 0u, 0u, 0u};   // hi half = K 16..31 pad
  }

  const __bf16* krow0 = kh + ((size_t)head * MKEY + l16) * HD;

  // --- pass A: row max of scores ---
  float mx[8];
  #pragma unroll
  for (int r = 0; r < 8; ++r) mx[r] = -1e30f;

  {
    const __bf16*   kp    = lo ? krow0 : zpad;
    const ptrdiff_t kstep = lo ? 16 * HD : 0;
    for (int m0 = 0; m0 < MKEY; m0 += 16) {
      BfVec bk;
      bk.q[0] = *(const uint4*)kp;
      bk.q[1] = *(const uint4*)(kp + 8);
      kp += kstep;
      v8f c = {};
      c = __builtin_amdgcn_wmma_f32_16x16x32_bf16(false, aq.v, false, bk.v,
                                                  (short)0, c, false, false);
      #pragma unroll
      for (int r = 0; r < 8; ++r) mx[r] = fmaxf(mx[r], c[r] * 0.25f);
    }
  }

  // reduce max across the 16 lanes of each half (rows r / r+8)
  #pragma unroll
  for (int r = 0; r < 8; ++r) {
    #pragma unroll
    for (int msk = 1; msk < 16; msk <<= 1)
      mx[r] = fmaxf(mx[r], __shfl_xor(mx[r], msk, 16));
  }

  // --- pass B: p = exp(s - mx); accumulate rowsum and X = P * V^T ---
  float sm[8];
  #pragma unroll
  for (int r = 0; r < 8; ++r) sm[r] = 0.f;
  v8f xacc = {};

  const __bf16*   kp2    = lo ? krow0 : zpad;
  const ptrdiff_t toff   = lo ? 16 * HD : 0;
  const ptrdiff_t kstep2 = lo ? 32 * HD : 0;

  for (int m0 = 0; m0 < MKEY; m0 += 32) {
    #pragma unroll
    for (int t = 0; t < 2; ++t) {
      const __bf16* kt = kp2 + (t ? toff : 0);
      BfVec bk;
      bk.q[0] = *(const uint4*)kt;
      bk.q[1] = *(const uint4*)(kt + 8);
      v8f c = {};
      c = __builtin_amdgcn_wmma_f32_16x16x32_bf16(false, aq.v, false, bk.v,
                                                  (short)0, c, false, false);
      #pragma unroll
      for (int r = 0; r < 8; ++r) {
        float p = __expf(c[r] * 0.25f - mx[r]);
        sm[r] += p;
        plds[r + (lo ? 0 : 8)][16 * t + l16] = (__bf16)p;
      }
    }
    kp2 += kstep2;

    // prob tile as A operand (16n x 32K, K = m local)
    BfVec pa;
    {
      const __bf16* pp = &plds[l16][lo ? 0 : 8];
      pa.q[0] = *(const uint4*)pp;        // cols base..base+7
      pa.q[1] = *(const uint4*)(pp + 16); // cols base+16..base+23
    }
    // V^T tile as B operand (32K=m x 16d)
    BfVec bv;
    {
      const __bf16* vp = vh + ((size_t)head * HD + l16) * MKEY + m0 + (lo ? 0 : 8);
      bv.q[0] = *(const uint4*)vp;
      bv.q[1] = *(const uint4*)(vp + 16);
    }
    xacc = __builtin_amdgcn_wmma_f32_16x16x32_bf16(false, pa.v, false, bv.v,
                                                   (short)0, xacc, false, false);
  }

  // reduce rowsum across the 16 lanes of each half, then normalize X
  float inv[8];
  #pragma unroll
  for (int r = 0; r < 8; ++r) {
    #pragma unroll
    for (int msk = 1; msk < 16; msk <<= 1)
      sm[r] += __shfl_xor(sm[r], msk, 16);
    inv[r] = 1.0f / sm[r];
  }

  // --- scatter X to conv layout: channel o = d*4 + head ---
  #pragma unroll
  for (int r = 0; r < 8; ++r) {
    int n = n0 + r + (lo ? 0 : 8);
    xout[((size_t)(l16 * NH + head)) * NPIX + n] = xacc[r] * inv[r];
  }
}

// ---------------------------------------------------------------------------
// z1 = fea_pixel + (Wm @ x + bm)
// ---------------------------------------------------------------------------
__global__ __launch_bounds__(256) void merge_kernel(
    const float* __restrict__ x, const float* __restrict__ Wm,
    const float* __restrict__ bm, const float* __restrict__ fpix,
    float* __restrict__ z1)
{
  int g = blockIdx.x * 256 + threadIdx.x;
  if (g >= DCH * NPIX) return;
  int n = g % NPIX;
  int o = g / NPIX;
  float acc = bm[o];
  #pragma unroll
  for (int c = 0; c < DCH; ++c)
    acc += Wm[o * DCH + c] * x[(size_t)c * NPIX + n];
  z1[g] = fpix[g] + acc;
}

// ---------------------------------------------------------------------------
// LayerNorm over the last axis (length NPIX) per channel row. 64 blocks.
// ---------------------------------------------------------------------------
__global__ __launch_bounds__(256) void ln_rows(
    const float* __restrict__ in, float* __restrict__ out)
{
  int o = blockIdx.x, tid = threadIdx.x;
  __shared__ float r1[256], r2[256];
  const float* row = in + (size_t)o * NPIX;
  float s1 = 0.f, s2 = 0.f;
  for (int n = tid; n < NPIX; n += 256) {
    float t = row[n];
    s1 += t; s2 += t * t;
  }
  r1[tid] = s1; r2[tid] = s2;
  __syncthreads();
  for (int s = 128; s > 0; s >>= 1) {
    if (tid < s) { r1[tid] += r1[tid + s]; r2[tid] += r2[tid + s]; }
    __syncthreads();
  }
  float mean = r1[0] * (1.0f / NPIX);
  float var  = r2[0] * (1.0f / NPIX) - mean * mean;
  float rs   = rsqrtf(var + 1e-5f);
  float* orow = out + (size_t)o * NPIX;
  for (int n = tid; n < NPIX; n += 256)
    orow[n] = (row[n] - mean) * rs;
}

// ---------------------------------------------------------------------------
// Per-pixel channel MLP: m = relu(W2 relu(W1 y + b1) + b2); z2 = fea + m
// ---------------------------------------------------------------------------
__global__ __launch_bounds__(256) void mlp_kernel(
    const float* __restrict__ fea, const float* __restrict__ W1,
    const float* __restrict__ b1, const float* __restrict__ W2,
    const float* __restrict__ b2, float* __restrict__ z2)
{
  int n = blockIdx.x * 256 + threadIdx.x;   // NPIX = 20 * 256
  float y[DCH], macc[DCH];
  #pragma unroll
  for (int d = 0; d < DCH; ++d) {
    y[d]    = fea[(size_t)d * NPIX + n];
    macc[d] = b2[d];
  }
  for (int o = 0; o < H2; ++o) {
    float h = b1[o];
    #pragma unroll
    for (int d = 0; d < DCH; ++d) h += W1[o * DCH + d] * y[d];
    h = fmaxf(h, 0.f);
    #pragma unroll
    for (int d = 0; d < DCH; ++d) macc[d] += W2[d * H2 + o] * h;
  }
  #pragma unroll
  for (int d = 0; d < DCH; ++d)
    z2[(size_t)d * NPIX + n] = y[d] + fmaxf(macc[d], 0.f);
}

// ---------------------------------------------------------------------------
extern "C" void kernel_launch(void* const* d_in, const int* in_sizes, int n_in,
                              void* d_out, int out_size, void* d_ws, size_t ws_size,
                              hipStream_t stream) {
  const float* fea_pixel    = (const float*)d_in[0];
  const float* fea_keypoint = (const float*)d_in[1];
  const float* Wq = (const float*)d_in[2];
  const float* bq = (const float*)d_in[3];
  const float* Wk = (const float*)d_in[4];
  const float* bk = (const float*)d_in[5];
  const float* Wv = (const float*)d_in[6];
  const float* bv = (const float*)d_in[7];
  const float* Wm = (const float*)d_in[8];
  const float* bm = (const float*)d_in[9];
  const float* W1 = (const float*)d_in[10];
  const float* b1 = (const float*)d_in[11];
  const float* W2 = (const float*)d_in[12];
  const float* b2 = (const float*)d_in[13];
  float* out = (float*)d_out;

  char* ws = (char*)d_ws;
  // bf16 buffers (dead after attn_flash)
  __bf16* qh = (__bf16*)(ws + 0);                    // 4*5120*16*2 = 655360
  __bf16* kh = (__bf16*)(ws + 655360);               // 4*4096*16*2 = 524288
  __bf16* vh = (__bf16*)(ws + 1179648);              // 524288
  float*  x   = (float*)(ws + 1703936);              // 64*5120*4 = 1310720
  float*  fea = (float*)(ws + 3014656);              // 1310720
  __bf16* zpad = (__bf16*)(ws + 4325376);            // 64 B zero pad
  // z1 / z2 alias the dead qh/kh/vh region (1703936 >= 1310720)
  float*  z1  = (float*)(ws + 0);
  float*  z2  = (float*)(ws + 0);

  // zero pad + projections -> bf16 per-head layouts
  init_zpad<<<1, 32, 0, stream>>>(zpad);
  proj_kernel<<<(DCH * NPIX + 255) / 256, 256, 0, stream>>>(
      fea_pixel, Wq, bq, qh, NPIX, 0);
  proj_kernel<<<(DCH * MKEY + 255) / 256, 256, 0, stream>>>(
      fea_keypoint, Wk, bk, kh, MKEY, 0);
  proj_kernel<<<(DCH * MKEY + 255) / 256, 256, 0, stream>>>(
      fea_keypoint, Wv, bv, vh, MKEY, 1);

  // flash attention: 4 heads * 320 n-tiles, one wave each
  attn_flash<<<NH * (NPIX / 16), 32, 0, stream>>>(qh, kh, vh, zpad, x);

  // merge projection + residual, then LN over N
  merge_kernel<<<(DCH * NPIX + 255) / 256, 256, 0, stream>>>(x, Wm, bm, fea_pixel, z1);
  ln_rows<<<DCH, 256, 0, stream>>>(z1, fea);

  // channel MLP + residual, then final LN over N into d_out
  mlp_kernel<<<NPIX / 256, 256, 0, stream>>>(fea, W1, b1, W2, b2, z2);
  ln_rows<<<DCH, 256, 0, stream>>>(z2, out);
}
